// MSTAGAT_Net_309237646182
// MI455X (gfx1250) — matrix operand order repaired and checked by reading, not compiled, not run on hardware
//
#include <hip/hip_runtime.h>
#include <hip/hip_bf16.h>
#include <math.h>

typedef __attribute__((ext_vector_type(16))) _Float16 v16h;
typedef __attribute__((ext_vector_type(8)))  float    v8f;

#define BB   16
#define TT   64
#define NN   5000
#define HIDD 32
#define FCC  16
#define HDS  4
#define RRK  8
#define HORZ 14

#define ROWS (BB*NN)       /* 80000 */
#define YCNT (ROWS*TT)     /* 5,120,000 */

/* ---- small workspace region (float indices), zeroed each launch ---- */
#define WS_SUMY   0
#define WS_SQY    1
#define WS_SUMZ   2
#define WS_SQZ    3
#define WS_A1     4
#define WS_B1     5
#define WS_ALC    16      /* 16: combined per-channel scale  */
#define WS_BEC    32      /* 16: combined per-channel bias   */
#define WS_WQKV   64      /* 3072: qkv_lw@qkv_hw [32x96]     */
#define WS_BQKV   3200    /* 96                              */
#define WS_WOUT   3328    /* 1024: out_lw@out_hw [32x32]     */
#define WS_BOUT   4352    /* 32                              */
#define WS_VPSUM  4400    /* 32: [H,R]                       */
#define WS_KV     4432    /* 4096: [B,H,8,8]                 */
#define WS_TMP    8528    /* 4096: [B,H,R,8]                 */
#define WS_DENOM  12624   /* 20000: [H,N]                    */
#define WS_SCST   32624   /* 192: per (s,o) sum/sumsq        */
#define WS_SCA    32816   /* 96                              */
#define WS_SCB    32912   /* 96                              */
#define WS_ALPHA  33008   /* 3: softmax(fusion_w)            */
#define WS_REG    33012

#define OFF_A_F   65536                    /* z1 (B*N x T), later h2   */
#define OFF_B_F   (OFF_A_F + YCNT)         /* h / outm / h_final       */
#define OFF_C_F   (OFF_B_F + ROWS*HIDD)    /* qkv (x96) / conv buffer  */

static __device__ __forceinline__ float elu1(float x) {
  return x > 0.f ? x + 1.f : __expf(x);
}

static __device__ __forceinline__ float blk_reduce(float v, float* sb) {
  int tid = threadIdx.x;
  __syncthreads();
  sb[tid] = v;
  __syncthreads();
  for (int s = blockDim.x >> 1; s > 0; s >>= 1) {
    if (tid < s) sb[tid] += sb[tid + s];
    __syncthreads();
  }
  return sb[0];
}

static __device__ __forceinline__ v8f wmma16(v16h a, v16h b, v8f c) {
  return __builtin_amdgcn_wmma_f32_16x16x32_f16(false, a, false, b, (short)0, c,
                                                false, false);
}

/* ===== K: precompute effective weights + softmax(alpha) ===== */
__global__ void precompute_k(const float* __restrict__ qlw, const float* __restrict__ qlb,
                             const float* __restrict__ qhw, const float* __restrict__ qhb,
                             const float* __restrict__ olw, const float* __restrict__ olb,
                             const float* __restrict__ ohw, const float* __restrict__ ohb,
                             const float* __restrict__ fw, float* __restrict__ wsf) {
  int tid = threadIdx.x;
  for (int idx = tid; idx < 32 * 96; idx += 256) {
    int i = idx / 96, j = idx % 96;
    float s = 0.f;
    for (int k = 0; k < 24; ++k) s += qlw[i * 24 + k] * qhw[k * 96 + j];
    wsf[WS_WQKV + idx] = s;
  }
  for (int j = tid; j < 96; j += 256) {
    float s = qhb[j];
    for (int k = 0; k < 24; ++k) s += qlb[k] * qhw[k * 96 + j];
    wsf[WS_BQKV + j] = s;
  }
  for (int idx = tid; idx < 32 * 32; idx += 256) {
    int i = idx / 32, j = idx % 32;
    float s = 0.f;
    for (int k = 0; k < 8; ++k) s += olw[i * 8 + k] * ohw[k * 32 + j];
    wsf[WS_WOUT + idx] = s;
  }
  for (int j = tid; j < 32; j += 256) {
    float s = ohb[j];
    for (int k = 0; k < 8; ++k) s += olb[k] * ohw[k * 32 + j];
    wsf[WS_BOUT + j] = s;
  }
  if (tid == 0) {
    float f0 = fw[0], f1 = fw[1], f2 = fw[2];
    float m = fmaxf(f0, fmaxf(f1, f2));
    float e0 = __expf(f0 - m), e1 = __expf(f1 - m), e2 = __expf(f2 - m);
    float s = e0 + e1 + e2;
    wsf[WS_ALPHA + 0] = e0 / s;
    wsf[WS_ALPHA + 1] = e1 / s;
    wsf[WS_ALPHA + 2] = e2 / s;
  }
}

/* ===== K1: depthwise temporal conv + global stats ===== */
__global__ void dwconv_k(const float* __restrict__ x, const float* __restrict__ dww,
                         const float* __restrict__ dwb, float* __restrict__ y1,
                         float* __restrict__ wsf) {
  int idx = blockIdx.x * 256 + threadIdx.x;
  float val = 0.f;
  if (idx < YCNT) {
    int r = idx >> 6, t = idx & 63;
    int b = r / NN, n = r % NN;
    const float* xb = x + (size_t)b * TT * NN + n;
    float xm = (t > 0) ? xb[(size_t)(t - 1) * NN] : 0.f;
    float x0 = xb[(size_t)t * NN];
    float xp = (t < TT - 1) ? xb[(size_t)(t + 1) * NN] : 0.f;
    val = dww[0] * xm + dww[1] * x0 + dww[2] * xp + dwb[0];
    y1[idx] = val;
  }
  __shared__ float sb[256];
  float s = blk_reduce(val, sb);
  float q = blk_reduce(val * val, sb);
  if (threadIdx.x == 0) {
    atomicAdd(&wsf[WS_SUMY], s);
    atomicAdd(&wsf[WS_SQY], q);
  }
}

__global__ void bn1_fin_k(const float* __restrict__ g, const float* __restrict__ b,
                          float* __restrict__ wsf) {
  if (threadIdx.x == 0) {
    float cnt = (float)YCNT;
    float m = wsf[WS_SUMY] / cnt;
    float v = wsf[WS_SQY] / cnt - m * m;
    float a = g[0] * rsqrtf(v + 1e-5f);
    wsf[WS_A1] = a;
    wsf[WS_B1] = b[0] - m * a;
  }
}

/* ===== K3: z1 = relu(bn1(y1)) in place + stats ===== */
__global__ void z1_k(float* __restrict__ y1, float* __restrict__ wsf) {
  int idx = blockIdx.x * 256 + threadIdx.x;
  float z = 0.f;
  float a1 = wsf[WS_A1], b1 = wsf[WS_B1];
  if (idx < YCNT) {
    float v = a1 * y1[idx] + b1;
    z = v > 0.f ? v : 0.f;
    y1[idx] = z;
  }
  __shared__ float sb[256];
  float s = blk_reduce(z, sb);
  float q = blk_reduce(z * z, sb);
  if (threadIdx.x == 0) {
    atomicAdd(&wsf[WS_SUMZ], s);
    atomicAdd(&wsf[WS_SQZ], q);
  }
}

/* pw-conv is 1x1 => BN2 stats are affine in z1 stats; fold to per-ch scale/bias */
__global__ void ch_consts_k(const float* __restrict__ pww, const float* __restrict__ pwb,
                            const float* __restrict__ g2, const float* __restrict__ b2,
                            float* __restrict__ wsf) {
  int c = threadIdx.x;
  if (c >= FCC) return;
  float cnt = (float)YCNT;
  float mz = wsf[WS_SUMZ] / cnt;
  float vz = wsf[WS_SQZ] / cnt - mz * mz;
  float m2 = pww[c] * mz + pwb[c];
  float v2 = pww[c] * pww[c] * vz;
  float a2 = g2[c] * rsqrtf(v2 + 1e-5f);
  float bb2 = b2[c] - m2 * a2;
  wsf[WS_ALC + c] = a2 * pww[c];
  wsf[WS_BEC + c] = a2 * pwb[c] + bb2;
}

/* ===== K5: WMMA projection GEMM  [80000 x 1024] @ [1024 x 32] ===== */
__global__ void proj_gemm_k(const float* __restrict__ z1, const float* __restrict__ wsf,
                            const float* __restrict__ projw, const float* __restrict__ projb,
                            float* __restrict__ h) {
  const int lane = threadIdx.x;
  const int mm = lane & 15, hi = lane >> 4;
  const int row0 = blockIdx.x * 16;
  const float* zrow = z1 + (size_t)(row0 + mm) * TT;
  v8f c0 = {}, c1 = {};
  for (int kk = 0; kk < FCC * TT; kk += 32) {
    const int ch = kk >> 6;           /* channel constant within a 32-step */
    const float al = wsf[WS_ALC + ch];
    const float be = wsf[WS_BEC + ch];
    const int tb = kk & 63;
    v16h a;
#pragma unroll
    for (int e = 0; e < 16; ++e) {
      const int k = e + (hi << 4);
      float v = al * zrow[tb + k] + be;
      a[e] = (_Float16)(v > 0.f ? v : 0.f);
    }
    v16h b0, b1;
#pragma unroll
    for (int e = 0; e < 16; ++e) {
      const int k = e + (hi << 4);
      b0[e] = (_Float16)projw[(size_t)(kk + k) * HIDD + mm];
      b1[e] = (_Float16)projw[(size_t)(kk + k) * HIDD + mm + 16];
    }
    c0 = wmma16(a, b0, c0);
    c1 = wmma16(a, b1, c1);
  }
#pragma unroll
  for (int e = 0; e < 8; ++e) {
    const int m = e + (hi << 3);
    h[(size_t)(row0 + m) * HIDD + mm] = c0[e] + projb[mm];
    h[(size_t)(row0 + m) * HIDD + mm + 16] = c1[e] + projb[mm + 16];
  }
}

/* ===== K6: WMMA qkv GEMM  [80000 x 32] @ [32 x 96] ===== */
__global__ void qkv_gemm_k(const float* __restrict__ h, const float* __restrict__ wsf,
                           float* __restrict__ qkv) {
  const int lane = threadIdx.x;
  const int mm = lane & 15, hi = lane >> 4;
  const int row0 = blockIdx.x * 16;
  v16h a;
#pragma unroll
  for (int e = 0; e < 16; ++e) {
    const int k = e + (hi << 4);
    a[e] = (_Float16)h[(size_t)(row0 + mm) * HIDD + k];
  }
  v8f acc[6];
#pragma unroll
  for (int jt = 0; jt < 6; ++jt) {
    v16h b;
#pragma unroll
    for (int e = 0; e < 16; ++e) {
      const int k = e + (hi << 4);
      b[e] = (_Float16)wsf[WS_WQKV + k * 96 + jt * 16 + mm];
    }
    v8f c = {};
    acc[jt] = wmma16(a, b, c);
  }
#pragma unroll
  for (int jt = 0; jt < 6; ++jt) {
#pragma unroll
    for (int e = 0; e < 8; ++e) {
      const int m = e + (hi << 3);
      const int j = jt * 16 + mm;
      qkv[(size_t)(row0 + m) * 96 + j] = acc[jt][e] + wsf[WS_BQKV + j];
    }
  }
}

/* ===== K7: kv / tmp / vpsum reductions, one block per (b,h) ===== */
__global__ void kv_tmp_k(const float* __restrict__ qkv, const float* __restrict__ vpar,
                         float* __restrict__ wsf) {
  int bh = blockIdx.x, b = bh >> 2, hh = bh & 3;
  int tid = threadIdx.x;          /* 64 threads, thread owns (d,e) */
  int d = tid >> 3, e = tid & 7;
  __shared__ float sk[64][8], sv[64][8], sp[64][8];
  float akv = 0.f, atm = 0.f, avs = 0.f;
  for (int n0 = 0; n0 < NN; n0 += 64) {
    int n = n0 + tid;
    if (n < NN) {
      const float* row = qkv + ((size_t)b * NN + n) * 96;
#pragma unroll
      for (int j = 0; j < 8; ++j) {
        sk[tid][j] = elu1(row[32 + hh * 8 + j]);
        sv[tid][j] = row[64 + hh * 8 + j];
        sp[tid][j] = elu1(vpar[((size_t)hh * 8 + j) * NN + n]);
      }
    } else {
#pragma unroll
      for (int j = 0; j < 8; ++j) { sk[tid][j] = 0.f; sv[tid][j] = 0.f; sp[tid][j] = 0.f; }
    }
    __syncthreads();
    for (int i = 0; i < 64; ++i) {
      akv += sk[i][d] * sv[i][e];
      atm += sp[i][d] * sv[i][e];
      if (e == 0) avs += sp[i][d];
    }
    __syncthreads();
  }
  wsf[WS_KV + bh * 64 + d * 8 + e] = akv;
  wsf[WS_TMP + bh * 64 + d * 8 + e] = atm;
  if (b == 0 && e == 0) wsf[WS_VPSUM + hh * 8 + d] = avs;
}

__global__ void denom_k(const float* __restrict__ u, float* __restrict__ wsf) {
  int idx = blockIdx.x * 256 + threadIdx.x;
  if (idx >= HDS * NN) return;
  int hh = idx / NN, n = idx % NN;
  float s = 0.f;
#pragma unroll
  for (int r = 0; r < 8; ++r)
    s += elu1(u[((size_t)hh * NN + n) * 8 + r]) * wsf[WS_VPSUM + hh * 8 + r];
  wsf[WS_DENOM + idx] = s;
}

/* ===== K9: linear attention apply + low-rank graph bias, merged heads ===== */
__global__ void attn_merge_k(const float* __restrict__ qkv, const float* __restrict__ u,
                             const float* __restrict__ wsf, float* __restrict__ outm) {
  int r = blockIdx.x * 256 + threadIdx.x;
  if (r >= ROWS) return;
  int b = r / NN, n = r % NN;
  const float* row = qkv + (size_t)r * 96;
  float o[32];
  for (int hh = 0; hh < HDS; ++hh) {
    float qp[8];
    float zs = 1e-8f;
#pragma unroll
    for (int d = 0; d < 8; ++d) {
      qp[d] = elu1(row[hh * 8 + d]);
      zs += elu1(row[32 + hh * 8 + d]);
    }
    float zi = 1.f / zs;
    const float* kvp = wsf + WS_KV + (b * 4 + hh) * 64;
    const float* tmp = wsf + WS_TMP + (b * 4 + hh) * 64;
    float up[8];
#pragma unroll
    for (int rr = 0; rr < 8; ++rr) up[rr] = elu1(u[((size_t)hh * NN + n) * 8 + rr]);
    float dn = wsf[WS_DENOM + hh * NN + n] + 1e-8f;
#pragma unroll
    for (int e = 0; e < 8; ++e) {
      float s = 0.f, g = 0.f;
#pragma unroll
      for (int d = 0; d < 8; ++d) s += qp[d] * kvp[d * 8 + e];
#pragma unroll
      for (int rr = 0; rr < 8; ++rr) g += up[rr] * tmp[rr * 8 + e];
      o[hh * 8 + e] = zi * s + g / dn;
    }
  }
#pragma unroll
  for (int j = 0; j < 32; ++j) outm[(size_t)r * 32 + j] = o[j];
}

/* ===== K10: WMMA out-projection GEMM [80000 x 32] @ [32 x 32] ===== */
__global__ void outproj_gemm_k(const float* __restrict__ outm, const float* __restrict__ wsf,
                               float* __restrict__ h2) {
  const int lane = threadIdx.x;
  const int mm = lane & 15, hi = lane >> 4;
  const int row0 = blockIdx.x * 16;
  v16h a;
#pragma unroll
  for (int e = 0; e < 16; ++e) {
    const int k = e + (hi << 4);
    a[e] = (_Float16)outm[(size_t)(row0 + mm) * HIDD + k];
  }
  v8f c0 = {}, c1 = {};
  v16h b0, b1;
#pragma unroll
  for (int e = 0; e < 16; ++e) {
    const int k = e + (hi << 4);
    b0[e] = (_Float16)wsf[WS_WOUT + k * 32 + mm];
    b1[e] = (_Float16)wsf[WS_WOUT + k * 32 + mm + 16];
  }
  c0 = wmma16(a, b0, c0);
  c1 = wmma16(a, b1, c1);
#pragma unroll
  for (int e = 0; e < 8; ++e) {
    const int m = e + (hi << 3);
    h2[(size_t)(row0 + m) * HIDD + mm] = c0[e] + wsf[WS_BOUT + mm];
    h2[(size_t)(row0 + m) * HIDD + mm + 16] = c1[e] + wsf[WS_BOUT + mm + 16];
  }
}

/* ===== K11: WMMA im2col dilated conv over node axis (K=96 = 3 taps x 32) ===== */
__global__ void scconv_k(const float* __restrict__ h2, const float* __restrict__ scw,
                         const float* __restrict__ scb, float* __restrict__ conv,
                         float* __restrict__ wsf) {
  const int s = blockIdx.z, b = blockIdx.y;
  const int n0 = blockIdx.x * 16;
  const int dil = 1 << s;
  const int lane = threadIdx.x;
  const int mm = lane & 15, hi = lane >> 4;
  const int myrow = n0 + mm;              /* A-frag row = node index */
  v8f c0 = {}, c1 = {};
#pragma unroll
  for (int k = 0; k < 3; ++k) {
    const int nm = myrow + (k - 1) * dil;
    const bool inb = (myrow < NN) && (nm >= 0) && (nm < NN);
    const float* hp = h2 + ((size_t)b * NN + (inb ? nm : 0)) * HIDD;
    v16h a;
#pragma unroll
    for (int e = 0; e < 16; ++e) {
      const int ki = e + (hi << 4);
      a[e] = inb ? (_Float16)hp[ki] : (_Float16)0.f;
    }
    v16h b0, b1;
#pragma unroll
    for (int e = 0; e < 16; ++e) {
      const int ki = e + (hi << 4);
      b0[e] = (_Float16)scw[(((size_t)s * 32 + mm) * 32 + ki) * 3 + k];
      b1[e] = (_Float16)scw[(((size_t)s * 32 + mm + 16) * 32 + ki) * 3 + k];
    }
    c0 = wmma16(a, b0, c0);
    c1 = wmma16(a, b1, c1);
  }
  const float bi0 = scb[s * 32 + mm], bi1 = scb[s * 32 + mm + 16];
  float s0 = 0.f, q0 = 0.f, s1 = 0.f, q1 = 0.f;
#pragma unroll
  for (int e = 0; e < 8; ++e) {
    const int m = e + (hi << 3);
    const int n = n0 + m;
    if (n < NN) {
      float v0 = c0[e] + bi0;
      float v1 = c1[e] + bi1;
      size_t base = (((size_t)s * BB + b) * NN + n) * HIDD;
      conv[base + mm] = v0;
      conv[base + mm + 16] = v1;
      s0 += v0; q0 += v0 * v0; s1 += v1; q1 += v1 * v1;
    }
  }
  atomicAdd(&wsf[WS_SCST + (s * 32 + mm) * 2], s0);
  atomicAdd(&wsf[WS_SCST + (s * 32 + mm) * 2 + 1], q0);
  atomicAdd(&wsf[WS_SCST + (s * 32 + mm + 16) * 2], s1);
  atomicAdd(&wsf[WS_SCST + (s * 32 + mm + 16) * 2 + 1], q1);
}

__global__ void sc_consts_k(const float* __restrict__ g, const float* __restrict__ b,
                            float* __restrict__ wsf) {
  int i = threadIdx.x;
  if (i >= 96) return;
  float cnt = (float)ROWS;
  float m = wsf[WS_SCST + i * 2] / cnt;
  float v = wsf[WS_SCST + i * 2 + 1] / cnt - m * m;
  float a = g[i] * rsqrtf(v + 1e-5f);
  wsf[WS_SCA + i] = a;
  wsf[WS_SCB + i] = b[i] - m * a;
}

/* ===== K13: scale fusion + flow bottleneck + residual LayerNorm ===== */
__global__ void fuse_ln_k(const float* __restrict__ conv, const float* __restrict__ wsf,
                          const float* __restrict__ flw, const float* __restrict__ flb,
                          const float* __restrict__ fhw, const float* __restrict__ fhb,
                          const float* __restrict__ tg, const float* __restrict__ tb,
                          float* __restrict__ hf) {
  int r = blockIdx.x * 256 + threadIdx.x;
  if (r >= ROWS) return;
  int b = r / NN, n = r % NN;
  float fu[32];
#pragma unroll
  for (int o = 0; o < 32; ++o) {
    float acc = 0.f;
#pragma unroll
    for (int s = 0; s < 3; ++s) {
      float cv = conv[(((size_t)s * BB + b) * NN + n) * HIDD + o];
      float v = wsf[WS_SCA + s * 32 + o] * cv + wsf[WS_SCB + s * 32 + o];
      acc += wsf[WS_ALPHA + s] * (v > 0.f ? v : 0.f);
    }
    fu[o] = acc;
  }
  float tl[8];
#pragma unroll
  for (int j = 0; j < 8; ++j) {
    float a = flb[j];
    for (int o = 0; o < 32; ++o) a += fu[o] * flw[o * 8 + j];
    tl[j] = a;
  }
  float t[32];
  float mean = 0.f;
#pragma unroll
  for (int o = 0; o < 32; ++o) {
    float a = fhb[o];
#pragma unroll
    for (int j = 0; j < 8; ++j) a += tl[j] * fhw[j * 32 + o];
    t[o] = a + fu[o];
    mean += t[o];
  }
  mean *= (1.f / 32.f);
  float var = 0.f;
#pragma unroll
  for (int o = 0; o < 32; ++o) { float d = t[o] - mean; var += d * d; }
  var *= (1.f / 32.f);
  float inv = rsqrtf(var + 1e-5f);
#pragma unroll
  for (int o = 0; o < 32; ++o)
    hf[(size_t)r * 32 + o] = tg[o] * (t[o] - mean) * inv + tb[o];
}

/* ===== K14: horizon predictor with exponential-decay refinement ===== */
__global__ void pred_k(const float* __restrict__ hf, const float* __restrict__ x,
                       const float* __restrict__ p1w, const float* __restrict__ p1b,
                       const float* __restrict__ plg, const float* __restrict__ plb,
                       const float* __restrict__ p2w, const float* __restrict__ p2b,
                       const float* __restrict__ g1w, const float* __restrict__ g1b,
                       const float* __restrict__ g2w, const float* __restrict__ g2b,
                       float* __restrict__ out) {
  int r = blockIdx.x * 256 + threadIdx.x;
  if (r >= ROWS) return;
  int b = r / NN, n = r % NN;
  const float* h = hf + (size_t)r * 32;
  float p1[8], g1[8];
#pragma unroll
  for (int j = 0; j < 8; ++j) {
    float a = p1b[j], g = g1b[j];
    for (int o = 0; o < 32; ++o) { a += h[o] * p1w[o * 8 + j]; g += h[o] * g1w[o * 8 + j]; }
    p1[j] = a;
    g1[j] = g > 0.f ? g : 0.f;
  }
  float m = 0.f;
#pragma unroll
  for (int j = 0; j < 8; ++j) m += p1[j];
  m *= 0.125f;
  float v = 0.f;
#pragma unroll
  for (int j = 0; j < 8; ++j) { float d = p1[j] - m; v += d * d; }
  v *= 0.125f;
  float inv = rsqrtf(v + 1e-5f);
  float pr[8];
#pragma unroll
  for (int j = 0; j < 8; ++j) {
    float t = plg[j] * (p1[j] - m) * inv + plb[j];
    pr[j] = t > 0.f ? t : 0.f;
  }
  float last = x[((size_t)b * TT + (TT - 1)) * NN + n];
#pragma unroll
  for (int kk = 0; kk < HORZ; ++kk) {
    float pd = p2b[kk], gt = g2b[kk];
#pragma unroll
    for (int j = 0; j < 8; ++j) { pd += pr[j] * p2w[j * HORZ + kk]; gt += g1[j] * g2w[j * HORZ + kk]; }
    float gate = 1.f / (1.f + __expf(-gt));
    float decay = __expf(-0.1f * (float)(kk + 1));
    out[(size_t)r * HORZ + kk] = gate * pd + (1.f - gate) * last * decay;
  }
}

/* ===== K15: reg = exp(log_reg) * mean(|u@v|) ===== */
__global__ void reg_k(const float* __restrict__ u, const float* __restrict__ v,
                      float* __restrict__ wsf) {
  int hn = blockIdx.x;
  int hh = hn / NN, n = hn % NN;
  float ur[8];
#pragma unroll
  for (int r = 0; r < 8; ++r) ur[r] = u[((size_t)hh * NN + n) * 8 + r];
  float s = 0.f;
  for (int m = threadIdx.x; m < NN; m += 256) {
    float d = 0.f;
#pragma unroll
    for (int r = 0; r < 8; ++r) d += ur[r] * v[((size_t)hh * 8 + r) * NN + m];
    s += fabsf(d);
  }
  __shared__ float sb[256];
  float tot = blk_reduce(s, sb);
  if (threadIdx.x == 0) atomicAdd(&wsf[WS_REG], tot);
}

__global__ void reg_fin_k(const float* __restrict__ lr, const float* __restrict__ wsf,
                          float* __restrict__ out) {
  if (threadIdx.x == 0)
    out[(size_t)ROWS * HORZ] = __expf(lr[0]) * wsf[WS_REG] / (4.f * 5000.f * 5000.f);
}

extern "C" void kernel_launch(void* const* d_in, const int* in_sizes, int n_in,
                              void* d_out, int out_size, void* d_ws, size_t ws_size,
                              hipStream_t stream) {
  (void)in_sizes; (void)n_in; (void)out_size; (void)ws_size;
  const float* x      = (const float*)d_in[0];
  const float* dw_w   = (const float*)d_in[1];
  const float* dw_b   = (const float*)d_in[2];
  const float* bn1_g  = (const float*)d_in[3];
  const float* bn1_b  = (const float*)d_in[4];
  const float* pw_w   = (const float*)d_in[5];
  const float* pw_b   = (const float*)d_in[6];
  const float* bn2_g  = (const float*)d_in[7];
  const float* bn2_b  = (const float*)d_in[8];
  const float* proj_w = (const float*)d_in[9];
  const float* proj_b = (const float*)d_in[10];
  const float* qkv_lw = (const float*)d_in[11];
  const float* qkv_lb = (const float*)d_in[12];
  const float* qkv_hw = (const float*)d_in[13];
  const float* qkv_hb = (const float*)d_in[14];
  const float* out_lw = (const float*)d_in[15];
  const float* out_lb = (const float*)d_in[16];
  const float* out_hw = (const float*)d_in[17];
  const float* out_hb = (const float*)d_in[18];
  const float* logreg = (const float*)d_in[19];
  const float* u      = (const float*)d_in[20];
  const float* v      = (const float*)d_in[21];
  const float* sc_w   = (const float*)d_in[22];
  const float* sc_b   = (const float*)d_in[23];
  const float* sbn_g  = (const float*)d_in[24];
  const float* sbn_b  = (const float*)d_in[25];
  const float* fus_w  = (const float*)d_in[26];
  const float* flow_w = (const float*)d_in[27];
  const float* flow_b = (const float*)d_in[28];
  const float* fh_w   = (const float*)d_in[29];
  const float* fh_b   = (const float*)d_in[30];
  const float* tln_g  = (const float*)d_in[31];
  const float* tln_b  = (const float*)d_in[32];
  const float* p1_w   = (const float*)d_in[33];
  const float* p1_b   = (const float*)d_in[34];
  const float* pln_g  = (const float*)d_in[35];
  const float* pln_b  = (const float*)d_in[36];
  const float* p2_w   = (const float*)d_in[37];
  const float* p2_b   = (const float*)d_in[38];
  const float* g1_w   = (const float*)d_in[39];
  const float* g1_b   = (const float*)d_in[40];
  const float* g2_w   = (const float*)d_in[41];
  const float* g2_b   = (const float*)d_in[42];

  float* wsf  = (float*)d_ws;
  float* regA = wsf + OFF_A_F;   /* z1 [80000x64] -> later h2 [80000x32] */
  float* regB = wsf + OFF_B_F;   /* h -> outm -> h_final [80000x32]      */
  float* regC = wsf + OFF_C_F;   /* qkv [80000x96] -> conv [3x16x5000x32] */
  float* out  = (float*)d_out;

  hipMemsetAsync(d_ws, 0, OFF_A_F * sizeof(float), stream);

  precompute_k<<<1, 256, 0, stream>>>(qkv_lw, qkv_lb, qkv_hw, qkv_hb,
                                      out_lw, out_lb, out_hw, out_hb, fus_w, wsf);

  const int gY = (YCNT + 255) / 256;          /* 20000 */
  dwconv_k<<<gY, 256, 0, stream>>>(x, dw_w, dw_b, regA, wsf);
  bn1_fin_k<<<1, 32, 0, stream>>>(bn1_g, bn1_b, wsf);
  z1_k<<<gY, 256, 0, stream>>>(regA, wsf);
  ch_consts_k<<<1, 32, 0, stream>>>(pw_w, pw_b, bn2_g, bn2_b, wsf);

  proj_gemm_k<<<ROWS / 16, 32, 0, stream>>>(regA, wsf, proj_w, proj_b, regB);
  qkv_gemm_k<<<ROWS / 16, 32, 0, stream>>>(regB, wsf, regC);

  kv_tmp_k<<<BB * HDS, 64, 0, stream>>>(regC, v, wsf);
  denom_k<<<(HDS * NN + 255) / 256, 256, 0, stream>>>(u, wsf);
  attn_merge_k<<<(ROWS + 255) / 256, 256, 0, stream>>>(regC, u, wsf, regB);

  outproj_gemm_k<<<ROWS / 16, 32, 0, stream>>>(regB, wsf, regA);

  dim3 scg((NN + 15) / 16, BB, 3);
  scconv_k<<<scg, 32, 0, stream>>>(regA, sc_w, sc_b, regC, wsf);
  sc_consts_k<<<1, 128, 0, stream>>>(sbn_g, sbn_b, wsf);

  fuse_ln_k<<<(ROWS + 255) / 256, 256, 0, stream>>>(regC, wsf, flow_w, flow_b,
                                                    fh_w, fh_b, tln_g, tln_b, regB);

  pred_k<<<(ROWS + 255) / 256, 256, 0, stream>>>(regB, x, p1_w, p1_b, pln_g, pln_b,
                                                 p2_w, p2_b, g1_w, g1_b, g2_w, g2_b, out);

  reg_k<<<HDS * NN, 256, 0, stream>>>(u, v, wsf);
  reg_fin_k<<<1, 32, 0, stream>>>(logreg, wsf, out);
}
